// UctransUnet_1013612281910
// MI455X (gfx1250) — matrix operand order repaired and checked by reading, not compiled, not run on hardware
//
#include <hip/hip_runtime.h>
#include <hip/hip_bf16.h>
#include <math.h>

// ---------------------------------------------------------------------------
// CDNA5 (gfx1250) implementation of the UCTransNet-style reference.
// Dense linear algebra (Q/K/V projections, Q^T*K scores, P*V^T context, Wo,
// fc1, fc2) runs through a double-buffered batched WMMA GEMM
// (v_wmma_f32_16x16x32_bf16); the 3x3 reconstruction conv is an
// implicit-im2col WMMA GEMM.
// ---------------------------------------------------------------------------

typedef __attribute__((ext_vector_type(16))) __bf16 v16bf;
typedef __attribute__((ext_vector_type(8)))  float  v8f;

__device__ __forceinline__ unsigned short f2bf(float f) {
  unsigned int u = __float_as_uint(f);
  unsigned int r = u + 0x7FFFu + ((u >> 16) & 1u);   // round-to-nearest-even
  return (unsigned short)(r >> 16);
}

// ---------------------------------------------------------------------------
// Batched GEMM: C[z] = alpha * op(A[z]) * op(B[z]) (+bias)(+res)(gelu?)
// Block tile 128x128, K-step 32, 8 waves (2 along M x 4 along N),
// each wave computes a 64x32 tile as 4x2 WMMA 16x16x32 bf16 MACs.
// Double-buffered LDS staging in WMMA fragment order.
// transA: A stored [K,M] (element (m,k) at A[k*lda+m])
// transB: B stored [N,K] (element (k,n) at B[n*ldb+k])
// Requires K % 32 == 0 (true for every GEMM in this model).
// ---------------------------------------------------------------------------
__global__ __launch_bounds__(256)
void gemm_bf16_wmma(const float* __restrict__ A, int lda, long strideA, int transA,
                    const float* __restrict__ Bm, int ldb, long strideB, int transB,
                    float* __restrict__ Cm, int ldc, long strideC,
                    const float* __restrict__ bias,
                    const float* __restrict__ res, int ldr, long strideR,
                    int M, int N, int K, float alpha, int gelu)
{
  // fragment-ordered LDS: [buf][16x32 tile][lane][16 bf16]
  __shared__ unsigned short lds_a[2][8][32][16];   // 16 KB
  __shared__ unsigned short lds_b[2][8][32][16];   // 16 KB

  const int z = blockIdx.z;
  A  += (long)z * strideA;
  Bm += (long)z * strideB;
  Cm += (long)z * strideC;
  if (res) res += (long)z * strideR;

  const int blockM = blockIdx.y * 128;
  const int blockN = blockIdx.x * 128;
  const int tid   = threadIdx.x;
  const int lane  = tid & 31;
  const int wave  = tid >> 5;
  const int waveM = wave >> 2;    // 0..1
  const int waveN = wave & 3;     // 0..3

  v8f acc[4][2] = {};

  // Stage one 128x32 A tile and one 32x128 B tile (bf16, fragment order).
  // ISA A layout (16-bit 16x32): lane = r + 16*((k>>3)&1), elem = (k&7)+8*(k>=16)
  // ISA B layout (16-bit 32x16): lane = n + 16*((k>>3)&1), elem = (k&7)+8*(k>=16)
  auto stage = [&](int buf, int kk) {
#pragma unroll
    for (int i = 0; i < 8; ++i) {
      int p   = tid * 8 + i;          // 0..2047 pair index
      int row = p >> 4;               // 0..127
      int kp  = (p & 15) << 1;        // even k 0..30
      int grow = blockM + row;
      int gk   = kk + kp;
      float f0 = 0.f, f1 = 0.f;
      if (grow < M) {
        if (!transA) {
          const float* ap = A + (long)grow * lda + gk;
          float2 t2 = *(const float2*)ap; f0 = t2.x; f1 = t2.y;
          if (i == 0 && gk + 64 < K) __builtin_prefetch(ap + 64, 0, 1);
        } else {
          f0 = A[(long)gk * lda + grow];
          f1 = A[(long)(gk + 1) * lda + grow];
        }
      }
      unsigned int pk = (unsigned int)f2bf(f0) | ((unsigned int)f2bf(f1) << 16);
      int h = (kp >> 3) & 1;
      int e = (kp & 7) + ((kp >= 16) ? 8 : 0);
      *(unsigned int*)&lds_a[buf][row >> 4][(row & 15) + 16 * h][e] = pk;
    }
#pragma unroll
    for (int i = 0; i < 8; ++i) {
      int p  = tid * 8 + i;
      int n  = p >> 4;                // 0..127
      int kp = (p & 15) << 1;
      int gn = blockN + n;
      int gk = kk + kp;
      float f0 = 0.f, f1 = 0.f;
      if (gn < N) {
        if (!transB) {
          f0 = Bm[(long)gk * ldb + gn];
          f1 = Bm[(long)(gk + 1) * ldb + gn];
        } else {
          const float* bp = Bm + (long)gn * ldb + gk;
          float2 t2 = *(const float2*)bp; f0 = t2.x; f1 = t2.y;
          if (i == 0 && gk + 64 < K) __builtin_prefetch(bp + 64, 0, 1);
        }
      }
      unsigned int pk = (unsigned int)f2bf(f0) | ((unsigned int)f2bf(f1) << 16);
      int h = (kp >> 3) & 1;
      int e = (kp & 7) + ((kp >= 16) ? 8 : 0);
      *(unsigned int*)&lds_b[buf][n >> 4][(n & 15) + 16 * h][e] = pk;
    }
  };

  stage(0, 0);
  int cur = 0;
  for (int kk = 0; kk < K; kk += 32) {
    __syncthreads();
    v16bf af[4], bfrag[2];
#pragma unroll
    for (int m = 0; m < 4; ++m)
      af[m] = *(const v16bf*)&lds_a[cur][waveM * 4 + m][lane][0];
#pragma unroll
    for (int n = 0; n < 2; ++n)
      bfrag[n] = *(const v16bf*)&lds_b[cur][waveN * 2 + n][lane][0];

    if (kk + 32 < K) stage(cur ^ 1, kk + 32);   // overlap next-tile staging

#pragma unroll
    for (int m = 0; m < 4; ++m)
#pragma unroll
      for (int n = 0; n < 2; ++n)
        acc[m][n] = __builtin_amdgcn_wmma_f32_16x16x32_bf16(
            false, af[m], false, bfrag[n], (short)0, acc[m][n], false, false);

    cur ^= 1;
  }

  // ---- epilogue: D layout (32-bit 16x16): row = v + 8*(lane>>4), col = lane&15 ----
  const int lrow = lane >> 4;
  const int lcol = lane & 15;
#pragma unroll
  for (int m = 0; m < 4; ++m) {
#pragma unroll
    for (int n = 0; n < 2; ++n) {
      int col = blockN + waveN * 32 + n * 16 + lcol;
      if (col >= N) continue;
      int rbase = blockM + waveM * 64 + m * 16 + lrow * 8;
#pragma unroll
      for (int v = 0; v < 8; ++v) {
        int r = rbase + v;
        if (r >= M) continue;
        float val = acc[m][n][v] * alpha;
        if (bias) val += bias[col];
        if (res)  val += res[(long)r * ldr + col];
        if (gelu) val = 0.5f * val * (1.0f + erff(val * 0.70710678118654752f));
        Cm[(long)r * ldc + col] = val;
      }
    }
  }
}

// ---------------------------------------------------------------------------
// 3x3 conv (pad 1, C->C) as implicit-im2col WMMA GEMM per batch image:
//   M = H*W spatial positions (multiple of 128), N = Cout, K = Cin*9.
// A(m,k) gathers the padded 3x3 neighborhood; B is rec_w viewed [Cout, Cin*9]
// (trans-B, contiguous rows). Output written channel-major [b, co, H, W].
// ---------------------------------------------------------------------------
__global__ __launch_bounds__(256)
void conv3x3_wmma(const float* __restrict__ x, const float* __restrict__ w,
                  const float* __restrict__ bias, float* __restrict__ y,
                  int C, int H)
{
  __shared__ unsigned short lds_a[2][8][32][16];
  __shared__ unsigned short lds_b[2][8][32][16];

  const int b  = blockIdx.z;
  const int HW = H * H;
  const float* xb = x + (long)b * C * HW;
  float* yb       = y + (long)b * C * HW;
  const int K = C * 9;

  const int blockM = blockIdx.y * 128;    // spatial
  const int blockN = blockIdx.x * 128;    // cout
  const int tid   = threadIdx.x;
  const int lane  = tid & 31;
  const int wave  = tid >> 5;
  const int waveM = wave >> 2;
  const int waveN = wave & 3;

  v8f acc[4][2] = {};

  auto gatherA = [&](int yy, int xx, int k) -> float {
    int ci = k / 9;
    int t  = k - ci * 9;
    int dy = t / 3 - 1;
    int dx = t - (t / 3) * 3 - 1;
    int sy = yy + dy, sx = xx + dx;
    return (sy >= 0 && sy < H && sx >= 0 && sx < H)
               ? xb[(long)ci * HW + sy * H + sx] : 0.f;
  };

  auto stage = [&](int buf, int kk) {
#pragma unroll
    for (int i = 0; i < 8; ++i) {
      int p   = tid * 8 + i;
      int row = p >> 4;
      int kp  = (p & 15) << 1;
      int m   = blockM + row;
      int gk  = kk + kp;
      float f0 = 0.f, f1 = 0.f;
      if (m < HW) {
        int yy = m / H, xx = m - (m / H) * H;
        f0 = gatherA(yy, xx, gk);
        f1 = gatherA(yy, xx, gk + 1);
      }
      unsigned int pk = (unsigned int)f2bf(f0) | ((unsigned int)f2bf(f1) << 16);
      int h = (kp >> 3) & 1;
      int e = (kp & 7) + ((kp >= 16) ? 8 : 0);
      *(unsigned int*)&lds_a[buf][row >> 4][(row & 15) + 16 * h][e] = pk;
    }
#pragma unroll
    for (int i = 0; i < 8; ++i) {
      int p  = tid * 8 + i;
      int n  = p >> 4;
      int kp = (p & 15) << 1;
      int gn = blockN + n;
      int gk = kk + kp;
      float f0 = 0.f, f1 = 0.f;
      if (gn < C) {
        const float* wp = w + (long)gn * K + gk;   // rec_w[co, ci, ky, kx] row
        float2 t2 = *(const float2*)wp; f0 = t2.x; f1 = t2.y;
        if (i == 0 && gk + 64 < K) __builtin_prefetch(wp + 64, 0, 1);
      }
      unsigned int pk = (unsigned int)f2bf(f0) | ((unsigned int)f2bf(f1) << 16);
      int h = (kp >> 3) & 1;
      int e = (kp & 7) + ((kp >= 16) ? 8 : 0);
      *(unsigned int*)&lds_b[buf][n >> 4][(n & 15) + 16 * h][e] = pk;
    }
  };

  stage(0, 0);
  int cur = 0;
  for (int kk = 0; kk < K; kk += 32) {
    __syncthreads();
    v16bf af[4], bfrag[2];
#pragma unroll
    for (int m = 0; m < 4; ++m)
      af[m] = *(const v16bf*)&lds_a[cur][waveM * 4 + m][lane][0];
#pragma unroll
    for (int n = 0; n < 2; ++n)
      bfrag[n] = *(const v16bf*)&lds_b[cur][waveN * 2 + n][lane][0];

    if (kk + 32 < K) stage(cur ^ 1, kk + 32);

#pragma unroll
    for (int m = 0; m < 4; ++m)
#pragma unroll
      for (int n = 0; n < 2; ++n)
        acc[m][n] = __builtin_amdgcn_wmma_f32_16x16x32_bf16(
            false, af[m], false, bfrag[n], (short)0, acc[m][n], false, false);

    cur ^= 1;
  }

  const int lrow = lane >> 4;
  const int lcol = lane & 15;
#pragma unroll
  for (int m = 0; m < 4; ++m) {
#pragma unroll
    for (int n = 0; n < 2; ++n) {
      int col = blockN + waveN * 32 + n * 16 + lcol;
      if (col >= C) continue;
      int rbase = blockM + waveM * 64 + m * 16 + lrow * 8;
      float bv = bias[col];
#pragma unroll
      for (int v = 0; v < 8; ++v) {
        int r = rbase + v;
        if (r >= HW) continue;
        yb[(long)col * HW + r] = acc[m][n][v] + bv;   // [b, co, y, x]
      }
    }
  }
}

// ---------------------------------------------------------------------------
// Patch embedding (non-overlapping conv == per-patch dot product).
// One block per (b, patch); patch cached in LDS; thread = output channel.
// ---------------------------------------------------------------------------
__global__ void patch_embed_kernel(const float* __restrict__ e, const float* __restrict__ w,
                                   const float* __restrict__ bias, const float* __restrict__ pos,
                                   float* __restrict__ out, int C, int p, int coff, int Himg)
{
  extern __shared__ float sm[];
  int bi = blockIdx.x;
  int b = bi >> 8;
  int n = bi & 255;
  int ph = n >> 4, pw = n & 15;
  int pp = p * p;
  int K = C * pp;
  for (int idx = threadIdx.x; idx < K; idx += blockDim.x) {
    int c = idx / pp;
    int r = idx - c * pp;
    int i = r / p, j = r - (r / p) * p;
    sm[idx] = e[(((long)b * C + c) * Himg + (ph * p + i)) * Himg + (pw * p + j)];
  }
  __syncthreads();
  int cout = threadIdx.x;      // blockDim.x == C
  float acc = 0.f;
  const float* wr = w + (long)cout * K;   // emb_w[cout, c, i, j] matches (c,i,j) order
  for (int k = 0; k < K; ++k) acc += sm[k] * wr[k];
  out[((long)b * 256 + n) * 960 + coff + cout] = acc + bias[cout] + pos[n * C + cout];
}

// ---------------------------------------------------------------------------
// Row LayerNorm: y = (x - mu) * rstd * w + b   (one block per row)
// ---------------------------------------------------------------------------
__global__ __launch_bounds__(256)
void ln_kernel(const float* __restrict__ x, int ldx,
               const float* __restrict__ w, const float* __restrict__ b,
               float* __restrict__ y, int ldy, int D, float eps)
{
  __shared__ float s1[256], s2[256];
  int row = blockIdx.x;
  const float* xr = x + (long)row * ldx;
  float sum = 0.f, sq = 0.f;
  for (int i = threadIdx.x; i < D; i += 256) { float v = xr[i]; sum += v; sq += v * v; }
  s1[threadIdx.x] = sum; s2[threadIdx.x] = sq;
  __syncthreads();
  for (int s = 128; s > 0; s >>= 1) {
    if (threadIdx.x < s) { s1[threadIdx.x] += s1[threadIdx.x + s]; s2[threadIdx.x] += s2[threadIdx.x + s]; }
    __syncthreads();
  }
  float mu  = s1[0] / (float)D;
  float var = s2[0] / (float)D - mu * mu;
  float rstd = rsqrtf(var + eps);
  float* yr = y + (long)row * ldy;
  for (int i = threadIdx.x; i < D; i += 256)
    yr[i] = (xr[i] - mu) * rstd * w[i] + b[i];
}

__global__ void copy_slice_kernel(const float* __restrict__ src, float* __restrict__ dst,
                                  int C, int coff)
{
  int idx = blockIdx.x * 256 + threadIdx.x;
  int total = 8 * 256 * C;
  if (idx >= total) return;
  int c = idx % C;
  int bn = idx / C;
  dst[idx] = src[(long)bn * 960 + coff + c];
}

// InstanceNorm stats over (C, KV) per (b,h): stats[2z]=mu, stats[2z+1]=rstd
__global__ __launch_bounds__(256)
void inorm_stats_kernel(const float* __restrict__ s, long CK, float* __restrict__ stats, float eps)
{
  __shared__ float s1[256], s2[256];
  long base = (long)blockIdx.x * CK;
  float sum = 0.f, sq = 0.f;
  for (long i = threadIdx.x; i < CK; i += 256) { float v = s[base + i]; sum += v; sq += v * v; }
  s1[threadIdx.x] = sum; s2[threadIdx.x] = sq;
  __syncthreads();
  for (int t = 128; t > 0; t >>= 1) {
    if (threadIdx.x < t) { s1[threadIdx.x] += s1[threadIdx.x + t]; s2[threadIdx.x] += s2[threadIdx.x + t]; }
    __syncthreads();
  }
  if (threadIdx.x == 0) {
    float mu  = s1[0] / (float)CK;
    float var = s2[0] / (float)CK - mu * mu;
    stats[2 * blockIdx.x]     = mu;
    stats[2 * blockIdx.x + 1] = rsqrtf(var + eps);
  }
}

// softmax over KV after instance normalization; one block per (z, c) row
__global__ __launch_bounds__(256)
void attn_softmax_kernel(float* __restrict__ s, const float* __restrict__ stats, int C, int KVd)
{
  __shared__ float red[256];
  int row = blockIdx.x;           // z*C + c
  int z = row / C;
  float mu = stats[2 * z], rstd = stats[2 * z + 1];
  float* sr = s + (long)row * KVd;
  float lv[4];
  int cnt = 0;
  float mx = -1e30f;
  for (int i = threadIdx.x; i < KVd; i += 256) {
    float v = (sr[i] - mu) * rstd;
    lv[cnt++] = v;
    mx = fmaxf(mx, v);
  }
  red[threadIdx.x] = mx; __syncthreads();
  for (int t = 128; t > 0; t >>= 1) {
    if (threadIdx.x < t) red[threadIdx.x] = fmaxf(red[threadIdx.x], red[threadIdx.x + t]);
    __syncthreads();
  }
  mx = red[0]; __syncthreads();
  float sum = 0.f; cnt = 0;
  for (int i = threadIdx.x; i < KVd; i += 256) { float ev = __expf(lv[cnt] - mx); lv[cnt] = ev; sum += ev; ++cnt; }
  red[threadIdx.x] = sum; __syncthreads();
  for (int t = 128; t > 0; t >>= 1) {
    if (threadIdx.x < t) red[threadIdx.x] += red[threadIdx.x + t];
    __syncthreads();
  }
  float inv = 1.0f / red[0];
  cnt = 0;
  for (int i = threadIdx.x; i < KVd; i += 256) sr[i] = lv[cnt++] * inv;
}

// mean over heads + transpose: out[b,n,c] = mean_h ctx[(h*8+b), c, n]
__global__ void head_mean_kernel(const float* __restrict__ ctx, float* __restrict__ out, int C)
{
  int idx = blockIdx.x * 256 + threadIdx.x;
  int total = 8 * 256 * C;
  if (idx >= total) return;
  int c = idx % C;
  int bn = idx / C;
  int b = bn >> 8;
  int n = bn & 255;
  float acc = 0.f;
  for (int h = 0; h < 4; ++h)
    acc += ctx[((long)(h * 8 + b) * C + c) * 256 + n];
  out[idx] = acc * 0.25f;
}

// bilinear upsample 16x16 -> (16S)x(16S), half-pixel centers, reading tokens [B,256,C]
__global__ void upsample_kernel(const float* __restrict__ enc, float* __restrict__ out, int C, int S)
{
  int H = 16 * S;
  long total = (long)8 * C * H * H;
  long idx = (long)blockIdx.x * 256 + threadIdx.x;
  if (idx >= total) return;
  int x = (int)(idx % H); long t = idx / H;
  int y = (int)(t % H); t /= H;
  int c = (int)(t % C); int b = (int)(t / C);
  float sx = (x + 0.5f) / (float)S - 0.5f;
  float sy = (y + 0.5f) / (float)S - 0.5f;
  int x0 = (int)floorf(sx), y0 = (int)floorf(sy);
  float wx = sx - (float)x0, wy = sy - (float)y0;
  int x1 = x0 + 1, y1 = y0 + 1;
  x0 = min(max(x0, 0), 15); x1 = min(max(x1, 0), 15);
  y0 = min(max(y0, 0), 15); y1 = min(max(y1, 0), 15);
  auto g = [&](int yy, int xx) { return enc[((long)b * 256 + yy * 16 + xx) * C + c]; };
  float v = (1.f - wy) * ((1.f - wx) * g(y0, x0) + wx * g(y0, x1)) +
            wy        * ((1.f - wx) * g(y1, x0) + wx * g(y1, x1));
  out[idx] = v;
}

// BatchNorm stats over (B,H,W) per channel
__global__ __launch_bounds__(256)
void bn_stats_kernel(const float* __restrict__ y, float* __restrict__ st, int C, int H, float eps)
{
  __shared__ float s1[256], s2[256];
  int c = blockIdx.x;
  long per = (long)H * H;
  long tot = 8 * per;
  float sum = 0.f, sq = 0.f;
  for (long i = threadIdx.x; i < tot; i += 256) {
    int b = (int)(i / per); long r = i % per;
    float v = y[((long)b * C + c) * per + r];
    sum += v; sq += v * v;
  }
  s1[threadIdx.x] = sum; s2[threadIdx.x] = sq;
  __syncthreads();
  for (int t = 128; t > 0; t >>= 1) {
    if (threadIdx.x < t) { s1[threadIdx.x] += s1[threadIdx.x + t]; s2[threadIdx.x] += s2[threadIdx.x + t]; }
    __syncthreads();
  }
  if (threadIdx.x == 0) {
    float mu  = s1[0] / (float)tot;
    float var = s2[0] / (float)tot - mu * mu;
    st[2 * c] = mu;
    st[2 * c + 1] = rsqrtf(var + eps);
  }
}

// in-place BN affine + ReLU + residual add of original input e
__global__ void bn_apply_kernel(float* __restrict__ y, const float* __restrict__ st,
                                const float* __restrict__ g, const float* __restrict__ bb,
                                const float* __restrict__ e, int C, int H)
{
  long total = (long)8 * C * H * H;
  long idx = (long)blockIdx.x * 256 + threadIdx.x;
  if (idx >= total) return;
  long per = (long)H * H;
  int c = (int)((idx / per) % C);
  float v = (y[idx] - st[2 * c]) * st[2 * c + 1] * g[c] + bb[c];
  v = fmaxf(v, 0.f);
  y[idx] = v + e[idx];
}

// ---------------------------------------------------------------------------
// Host orchestration
// ---------------------------------------------------------------------------
static inline void launch_gemm(hipStream_t st,
                               const float* A, int lda, long sA, int tA,
                               const float* B, int ldb, long sB, int tB,
                               float* C, int ldc, long sC,
                               const float* bias, const float* res, int ldr, long sR,
                               int M, int N, int K, float alpha, int gelu, int batch)
{
  dim3 g((unsigned)((N + 127) / 128), (unsigned)((M + 127) / 128), (unsigned)batch);
  gemm_bf16_wmma<<<g, dim3(256), 0, st>>>(A, lda, sA, tA, B, ldb, sB, tB, C, ldc, sC,
                                          bias, res, ldr, sR, M, N, K, alpha, gelu);
}

extern "C" void kernel_launch(void* const* d_in, const int* in_sizes, int n_in,
                              void* d_out, int out_size, void* d_ws, size_t ws_size,
                              hipStream_t stream)
{
  (void)in_sizes; (void)n_in; (void)out_size; (void)ws_size;

  static const int CHs[4]   = {64, 128, 256, 512};
  static const int PATCHs[4]= {16, 8, 4, 2};
  static const int COFF[4]  = {0, 64, 192, 448};
  static const int HIMG[4]  = {256, 128, 64, 32};

  const float* e_in[4] = {(const float*)d_in[0], (const float*)d_in[1],
                          (const float*)d_in[2], (const float*)d_in[3]};

  struct Layer { const float *ln_w, *ln_b, *lna_w, *lna_b, *Wq, *Wk, *Wv, *Wo,
                             *f1w, *f1b, *f2w, *f2b; };
  struct SP { const float *emb_w, *emb_b, *pos; Layer l[4];
              const float *fln_w, *fln_b, *rec_w, *rec_b, *bn_g, *bn_b; } sp[4];

  int idx = 4;  // inputs flattened depth-first in dict insertion order
  for (int s = 0; s < 4; ++s) {
    sp[s].emb_w = (const float*)d_in[idx++];
    sp[s].emb_b = (const float*)d_in[idx++];
    sp[s].pos   = (const float*)d_in[idx++];
    for (int li = 0; li < 4; ++li) {
      Layer& L = sp[s].l[li];
      L.ln_w  = (const float*)d_in[idx++];  L.ln_b  = (const float*)d_in[idx++];
      L.lna_w = (const float*)d_in[idx++];  L.lna_b = (const float*)d_in[idx++];
      L.Wq    = (const float*)d_in[idx++];  L.Wk    = (const float*)d_in[idx++];
      L.Wv    = (const float*)d_in[idx++];  L.Wo    = (const float*)d_in[idx++];
      L.f1w   = (const float*)d_in[idx++];  L.f1b   = (const float*)d_in[idx++];
      L.f2w   = (const float*)d_in[idx++];  L.f2b   = (const float*)d_in[idx++];
    }
    sp[s].fln_w = (const float*)d_in[idx++];  sp[s].fln_b = (const float*)d_in[idx++];
    sp[s].rec_w = (const float*)d_in[idx++];  sp[s].rec_b = (const float*)d_in[idx++];
    sp[s].bn_g  = (const float*)d_in[idx++];  sp[s].bn_b  = (const float*)d_in[idx++];
  }

  // ---- workspace layout (floats) ----
  float* W = (float*)d_ws;
  long off = 0;
  auto alloc = [&](long n) { float* p = W + off; off += n; return p; };
  float* emb_all = alloc(1966080L);        // [8,256,960]
  float* oa      = alloc(1966080L);        // ln(emb_all)
  float* emb_cur = alloc(1048576L);        // [8,256,C] running embedding
  float* o1      = alloc(1048576L);
  float* o2buf   = alloc(1048576L);
  float* encb    = alloc(1048576L);
  float* cmean   = alloc(1048576L);
  float* stats   = alloc(64L);             // per-(b,h) inorm stats
  float* bnst    = alloc(1024L);           // per-channel bn stats
  long bigOff = off;
  float* Qb = alloc(4194304L);             // [H, 2048, C]
  float* Kb = alloc(7864320L);             // [H, 2048, 960]
  float* Vb = alloc(7864320L);
  float* Sc = alloc(15728640L);            // [32, C, 960] scores/probs
  float* Cx = alloc(4194304L);             // [32, C, 256] context
  float* Mh = alloc(4194304L);             // [2048, 4C] mlp hidden
  float* resized = W + bigOff;             // aliases attention scratch (disjoint phase)

  float* outp = (float*)d_out;
  const long outoff[4] = {0L, 33554432L, 33554432L + 16777216L,
                          33554432L + 16777216L + 8388608L};

  const float inv_sqrt_kv = 0.03227486121839514f;  // 1/sqrt(960)

  // ---- phase 1: patch embeddings into emb_all[:, :, coff:coff+C] ----
  for (int s = 0; s < 4; ++s) {
    int C = CHs[s], p = PATCHs[s];
    size_t shmem = (size_t)C * p * p * sizeof(float);   // <= 64KB
    patch_embed_kernel<<<dim3(8 * 256), dim3(C), shmem, stream>>>(
        e_in[s], sp[s].emb_w, sp[s].emb_b, sp[s].pos, emb_all, C, p, COFF[s], HIMG[s]);
  }

  // ---- phase 2: per-scale CCT stacks + reconstruction ----
  for (int s = 0; s < 4; ++s) {
    int C = CHs[s];
    copy_slice_kernel<<<dim3((8 * 256 * C + 255) / 256), dim3(256), 0, stream>>>(
        emb_all, emb_cur, C, COFF[s]);

    for (int li = 0; li < 4; ++li) {
      const Layer& L = sp[s].l[li];
      ln_kernel<<<dim3(2048), dim3(256), 0, stream>>>(emb_cur, C, L.ln_w, L.ln_b, o1, C, C, 1e-5f);
      ln_kernel<<<dim3(2048), dim3(256), 0, stream>>>(emb_all, 960, L.lna_w, L.lna_b, oa, 960, 960, 1e-5f);

      // Q[h] = o1 @ Wq[h]            M=2048 N=C K=C, batch=H
      launch_gemm(stream, o1, C, 0L, 0, L.Wq, C, (long)C * C, 0,
                  Qb, C, 2048L * C, nullptr, nullptr, 0, 0L, 2048, C, C, 1.f, 0, 4);
      // K[h] = oa @ Wk[h], V[h] = oa @ Wv[h]   M=2048 N=960 K=960, batch=H
      launch_gemm(stream, oa, 960, 0L, 0, L.Wk, 960, 960L * 960, 0,
                  Kb, 960, 2048L * 960, nullptr, nullptr, 0, 0L, 2048, 960, 960, 1.f, 0, 4);
      launch_gemm(stream, oa, 960, 0L, 0, L.Wv, 960, 960L * 960, 0,
                  Vb, 960, 2048L * 960, nullptr, nullptr, 0, 0L, 2048, 960, 960, 1.f, 0, 4);

      // scores[z=(h*8+b)] = (Q_z)^T @ K_z / sqrt(KV)   M=C N=960 K=256
      launch_gemm(stream, Qb, C, 256L * C, 1, Kb, 960, 256L * 960, 0,
                  Sc, 960, (long)C * 960, nullptr, nullptr, 0, 0L,
                  C, 960, 256, inv_sqrt_kv, 0, 32);

      inorm_stats_kernel<<<dim3(32), dim3(256), 0, stream>>>(Sc, (long)C * 960, stats, 1e-5f);
      attn_softmax_kernel<<<dim3(32 * C), dim3(256), 0, stream>>>(Sc, stats, C, 960);

      // ctx[z] = probs_z @ (V_z)^T    M=C N=256 K=960
      launch_gemm(stream, Sc, 960, (long)C * 960, 0, Vb, 960, 256L * 960, 1,
                  Cx, 256, (long)C * 256, nullptr, nullptr, 0, 0L, C, 256, 960, 1.f, 0, 32);

      head_mean_kernel<<<dim3((8 * 256 * C + 255) / 256), dim3(256), 0, stream>>>(Cx, cmean, C);

      // o2 = cmean @ Wo + o1
      launch_gemm(stream, cmean, C, 0L, 0, L.Wo, C, 0L, 0,
                  o2buf, C, 0L, nullptr, o1, C, 0L, 2048, C, C, 1.f, 0, 1);
      // h = ln(o2)  (reuse o1)
      ln_kernel<<<dim3(2048), dim3(256), 0, stream>>>(o2buf, C, L.ln_w, L.ln_b, o1, C, C, 1e-5f);
      // mlph = gelu(h @ fc1 + b1)
      launch_gemm(stream, o1, C, 0L, 0, L.f1w, 4 * C, 0L, 0,
                  Mh, 4 * C, 0L, L.f1b, nullptr, 0, 0L, 2048, 4 * C, C, 1.f, 1, 1);
      // emb_cur = mlph @ fc2 + b2 + o2
      launch_gemm(stream, Mh, 4 * C, 0L, 0, L.f2w, C, 0L, 0,
                  emb_cur, C, 0L, L.f2b, o2buf, C, 0L, 2048, C, 4 * C, 1.f, 0, 1);
    }

    ln_kernel<<<dim3(2048), dim3(256), 0, stream>>>(emb_cur, C, sp[s].fln_w, sp[s].fln_b,
                                                    encb, C, C, 1e-6f);

    // ---- reconstruction ----
    int p = PATCHs[s];
    int H = 16 * p;
    int HW = H * H;
    long tot = (long)8 * C * HW;
    unsigned nb = (unsigned)((tot + 255) / 256);
    upsample_kernel<<<dim3(nb), dim3(256), 0, stream>>>(encb, resized, C, p);
    float* yout = outp + outoff[s];
    dim3 cg((unsigned)((C + 127) / 128), (unsigned)(HW / 128), 8);
    conv3x3_wmma<<<cg, dim3(256), 0, stream>>>(resized, sp[s].rec_w, sp[s].rec_b, yout, C, H);
    bn_stats_kernel<<<dim3(C), dim3(256), 0, stream>>>(yout, bnst, C, H, 1e-5f);
    bn_apply_kernel<<<dim3(nb), dim3(256), 0, stream>>>(yout, bnst, sp[s].bn_g, sp[s].bn_b,
                                                        e_in[s], C, H);
  }
}
